// BasicBlockBase_25580825215702
// MI455X (gfx1250) — matrix operand order repaired and checked by reading, not compiled, not run on hardware
//
#include <hip/hip_runtime.h>

// Problem constants (match reference)
#define NPTS 1000000
#define KOFF 27
#define CCH  32
#define BN_EPS 1e-5f
#define WFRAG_ELEMS (KOFF * 32 * 32)          // 27648 bf16 fragment elements
#define WFRAG_UINT4 (WFRAG_ELEMS * 2 / 16)    // 3456 uint4 = 54 KB

typedef __attribute__((ext_vector_type(16))) __bf16    v16bf;
typedef __attribute__((ext_vector_type(8)))  float     v8f;
typedef __attribute__((ext_vector_type(8)))  unsigned  v8u;

__device__ __forceinline__ unsigned short f2bf_bits(float f) {
  // round-to-nearest-even f32 -> bf16 (bit pattern)
  unsigned u = __builtin_bit_cast(unsigned, f);
  unsigned r = u + 0x7FFFu + ((u >> 16) & 1u);
  return (unsigned short)(r >> 16);
}

// ---------------------------------------------------------------------------
// One-time: pack W[k] (Cin x Cout fp32, row-major) into the per-lane bf16
// B-fragment layout consumed by v_wmma_f32_16x16x32_bf16:
//   flat i = (k*32 + lane)*32 + e ; lane L holds col N = L%16 (+16 for e>=16),
//   element e holds K = (e%16) + (L/16)*16.
// ---------------------------------------------------------------------------
__global__ __launch_bounds__(256) void pack_weights(
    const float* __restrict__ Wg, unsigned short* __restrict__ Wp) {
  int i = blockIdx.x * 256 + threadIdx.x;
  if (i < WFRAG_ELEMS) {
    int k    = i >> 10;       // / 1024
    int r    = i & 1023;
    int lane = r >> 5;
    int e    = r & 31;
    int g    = lane >> 4;
    int n, kk;
    if (e < 16) { n = (lane & 15);      kk = e + g * 16; }        // frag0
    else        { n = 16 + (lane & 15); kk = (e - 16) + g * 16; } // frag1
    Wp[i] = f2bf_bits(Wg[k * (CCH * CCH) + kk * CCH + n]);
  }
}

// ---------------------------------------------------------------------------
// Elementwise f32 -> packed bf16 (2 per dword). AFFINE: y = relu(x*s + b)
// per channel (fuses BN+ReLU of the previous stage into the repack).
// ---------------------------------------------------------------------------
template <bool AFFINE>
__global__ __launch_bounds__(256) void pack_bf16(
    const float* __restrict__ src, const float* __restrict__ scale,
    const float* __restrict__ bias, unsigned* __restrict__ dst,
    long long total_pairs) {
  long long i = (long long)blockIdx.x * 256 + threadIdx.x;
  if (i < total_pairs) {
    long long j = 2 * i;
    float a = src[j], b = src[j + 1];
    if constexpr (AFFINE) {
      int c0 = (int)(j & (CCH - 1));
      a = fmaxf(a * scale[c0] + bias[c0], 0.f);
      b = fmaxf(b * scale[c0 + 1] + bias[c0 + 1], 0.f);
    }
    dst[i] = ((unsigned)f2bf_bits(b) << 16) | (unsigned)f2bf_bits(a);
  }
}

// ---------------------------------------------------------------------------
// Gather + GEMM sparse conv on pre-packed bf16 features. One wave = 16 points.
//   A (16x32 bf16): lane L holds row M = L%16; half = L/16 selects
//     K in {half*16 .. half*16+15}  (contiguous, same k order as packed B)
//   B fragments: pre-packed 54 KB buffer, block-copied global -> LDS.
//   C (16x16 f32): lane L holds col N = L%16, rows M = v + (L/16)*8
// Mask applied as bitwise AND on packed bf16 dwords (exact for {0,1}).
// ---------------------------------------------------------------------------
__global__ __launch_bounds__(128) void sconv_wmma_bf16(
    const unsigned* __restrict__ xbf,  // [N][16] dwords = [N][32] bf16
    const int* __restrict__ nbr, const int* __restrict__ mask,
    const uint4* __restrict__ Wp,      // pre-packed B fragments (54 KB)
    float* __restrict__ hout, float* __restrict__ gsum,
    float* __restrict__ gsumsq) {
  __shared__ __align__(16) __bf16 sW[WFRAG_ELEMS];  // [k][lane][32 bf16]
  __shared__ float redS[CCH];
  __shared__ float redQ[CCH];

  const int tid = threadIdx.x;
  if (tid < CCH) { redS[tid] = 0.f; redQ[tid] = 0.f; }

  // Coalesced 54 KB block copy: global (L2-resident) -> LDS.
  uint4* s4 = (uint4*)sW;
#pragma unroll 4
  for (int i = tid; i < WFRAG_UINT4; i += 128) s4[i] = Wp[i];
  __syncthreads();

  const int wave = tid >> 5;
  const int lane = tid & 31;
  const int row  = lane & 15;  // A row / C column
  const int half = lane >> 4;
  const long tile = (long)blockIdx.x * 4 + wave;
  const long p    = tile * 16 + row;  // point gathered by this lane

  v8f acc0 = {};  // Cout 0..15
  v8f acc1 = {};  // Cout 16..31

  const long pK = p * KOFF;
  for (int k = 0; k < KOFF; ++k) {
    const int idx = nbr[pK + k];
    const unsigned mb = mask[pK + k] ? 0xFFFFFFFFu : 0u;
    const unsigned* xr = xbf + (long)idx * 16;  // 16 dwords / row
    // bf16 K {half*16..+15} -> dwords [half*8..+7]
    uint4 c0 = *(const uint4*)(xr + half * 8);
    uint4 c1 = *(const uint4*)(xr + half * 8 + 4);
    v8u au;
    au[0] = c0.x & mb; au[1] = c0.y & mb; au[2] = c0.z & mb; au[3] = c0.w & mb;
    au[4] = c1.x & mb; au[5] = c1.y & mb; au[6] = c1.z & mb; au[7] = c1.w & mb;
    v16bf a = __builtin_bit_cast(v16bf, au);

    const v16bf* bp = (const v16bf*)&sW[(k * 32 + lane) * 32];
    v16bf wb0 = bp[0];
    v16bf wb1 = bp[1];
    acc0 = __builtin_amdgcn_wmma_f32_16x16x32_bf16(false, a, false, wb0,
                                                   (short)0, acc0, false, false);
    acc1 = __builtin_amdgcn_wmma_f32_16x16x32_bf16(false, a, false, wb1,
                                                   (short)0, acc1, false, false);
  }

  // Write raw conv output + accumulate per-channel sum / sum-of-squares.
  const long m0 = tile * 16;
  float sA = 0.f, qA = 0.f, sB = 0.f, qB = 0.f;
#pragma unroll
  for (int v = 0; v < 8; ++v) {
    const long m = m0 + half * 8 + v;
    float a0 = acc0[v], a1 = acc1[v];
    hout[m * CCH + row]      = a0;
    hout[m * CCH + 16 + row] = a1;
    sA += a0; qA += a0 * a0;
    sB += a1; qB += a1 * a1;
  }
  atomicAdd(&redS[row], sA);       atomicAdd(&redQ[row], qA);
  atomicAdd(&redS[16 + row], sB);  atomicAdd(&redQ[16 + row], qB);
  __syncthreads();
  if (tid < CCH) {
    atomicAdd(&gsum[tid], redS[tid]);
    atomicAdd(&gsumsq[tid], redQ[tid]);
  }
}

// Fold BN stats into per-channel affine: y = h*scale + bias
__global__ void bn_finalize(const float* __restrict__ gsum,
                            const float* __restrict__ gsumsq,
                            const float* __restrict__ gamma,
                            const float* __restrict__ beta,
                            float* __restrict__ scale,
                            float* __restrict__ bias) {
  int c = threadIdx.x;
  if (c < CCH) {
    float inv_n = 1.0f / (float)NPTS;
    float mean  = gsum[c] * inv_n;
    float var   = gsumsq[c] * inv_n - mean * mean;
    float s     = gamma[c] * rsqrtf(var + BN_EPS);
    scale[c] = s;
    bias[c]  = beta[c] - mean * s;
  }
}

// out = relu(bn2(h2) + x)
__global__ __launch_bounds__(256) void residual_relu(
    const float* __restrict__ h2, const float* __restrict__ x,
    const float* __restrict__ scale, const float* __restrict__ bias,
    float* __restrict__ out, long long total) {
  long long i = (long long)blockIdx.x * 256 + threadIdx.x;
  if (i < total) {
    int c = (int)(i & (CCH - 1));
    float v = h2[i] * scale[c] + bias[c] + x[i];
    out[i] = fmaxf(v, 0.f);
  }
}

extern "C" void kernel_launch(void* const* d_in, const int* in_sizes, int n_in,
                              void* d_out, int out_size, void* d_ws,
                              size_t ws_size, hipStream_t stream) {
  const float* x     = (const float*)d_in[0];
  const int*   nbr1  = (const int*)d_in[1];
  const int*   mask1 = (const int*)d_in[2];
  const int*   nbr2  = (const int*)d_in[3];
  const int*   mask2 = (const int*)d_in[4];
  const float* W1    = (const float*)d_in[5];
  const float* W2    = (const float*)d_in[6];
  const float* g1    = (const float*)d_in[7];
  const float* b1    = (const float*)d_in[8];
  const float* g2    = (const float*)d_in[9];
  const float* b2    = (const float*)d_in[10];
  float* out = (float*)d_out;

  // Workspace: [stats 256f][pad 4KB][h1 128MB][h2 128MB][fbf 64MB][wp1][wp2]
  float* stats  = (float*)d_ws;
  float* sum1   = stats +   0;
  float* sumsq1 = stats +  32;
  float* scale1 = stats +  64;
  float* bias1  = stats +  96;
  float* sum2   = stats + 128;
  float* sumsq2 = stats + 160;
  float* scale2 = stats + 192;
  float* bias2  = stats + 224;
  float*    h1  = (float*)((char*)d_ws + 4096);
  float*    h2  = h1 + (size_t)NPTS * CCH;
  unsigned* fbf = (unsigned*)(h2 + (size_t)NPTS * CCH);  // bf16 features (reused)
  unsigned short* wp1 = (unsigned short*)(fbf + (size_t)NPTS * (CCH / 2));
  unsigned short* wp2 = wp1 + WFRAG_ELEMS;

  hipMemsetAsync(d_ws, 0, 1024, stream);  // zero stat accumulators

  const long long pairs = (long long)NPTS * (CCH / 2);  // 16M packed dwords
  const int pack_grid = (int)((pairs + 255) / 256);
  const int conv_grid = NPTS / 64;  // 4 waves/WG * 16 points/wave
  const int wpk_grid  = (WFRAG_ELEMS + 255) / 256;

  // One-time weight fragment packing (54 KB each, stays in L2).
  pack_weights<<<wpk_grid, 256, 0, stream>>>(W1, wp1);
  pack_weights<<<wpk_grid, 256, 0, stream>>>(W2, wp2);

  // x -> bf16
  pack_bf16<false><<<pack_grid, 256, 0, stream>>>(x, nullptr, nullptr, fbf, pairs);
  // conv1: gather(fbf) @ W1 -> h1 (raw) + stats1
  sconv_wmma_bf16<<<conv_grid, 128, 0, stream>>>(fbf, nbr1, mask1,
                                                 (const uint4*)wp1, h1,
                                                 sum1, sumsq1);
  bn_finalize<<<1, 32, 0, stream>>>(sum1, sumsq1, g1, b1, scale1, bias1);
  // relu(bn1(h1)) -> bf16 (reuse fbf)
  pack_bf16<true><<<pack_grid, 256, 0, stream>>>(h1, scale1, bias1, fbf, pairs);
  // conv2: gather(fbf) @ W2 -> h2 (raw) + stats2
  sconv_wmma_bf16<<<conv_grid, 128, 0, stream>>>(fbf, nbr2, mask2,
                                                 (const uint4*)wp2, h2,
                                                 sum2, sumsq2);
  bn_finalize<<<1, 32, 0, stream>>>(sum2, sumsq2, g2, b2, scale2, bias2);
  // out = relu(bn2(h2) + x)
  const long long total = (long long)NPTS * CCH;
  residual_relu<<<(int)((total + 255) / 256), 256, 0, stream>>>(
      h2, x, scale2, bias2, out, total);
}
